// ScoreAttention_16758962389372
// MI455X (gfx1250) — matrix operand order repaired and checked by reading, not compiled, not run on hardware
//
#include <hip/hip_runtime.h>
#include <hip/hip_bf16.h>

#define CH     256
#define NHEAD  8
#define HDIM   32
#define NGROUP 32
#define BATCH  2
#define HWTOK  4096
#define GNEPS  1e-5f

typedef __attribute__((ext_vector_type(16))) _Float16 v16h;
typedef __attribute__((ext_vector_type(8)))  float    v8f;
typedef __attribute__((ext_vector_type(4)))  int      v4i;
typedef __attribute__((ext_vector_type(4)))  unsigned u32x4;
typedef __attribute__((ext_vector_type(8)))  int      i32x8;

union F16Frag { v16h v; v4i q[2]; };

// ---------------------------------------------------------------------------
// WMMA fragment loaders (CDNA5 ISA 7.12.2 layouts, wave32)
// A 16x32 f16: lane<16 -> M=lane, VGPR0..3 K=0..7, VGPR4..7 K=16..23
//              lane>=16 -> same M, K+8 / K+24. Two contiguous b128 per lane.
__device__ __forceinline__ v16h load_a_frag(const _Float16* __restrict__ base,
                                            int row0, int ld, int k0) {
  const int lane = threadIdx.x & 31;
  const int hi   = lane >> 4;
  const int r    = lane & 15;
  const _Float16* p = base + (size_t)(row0 + r) * ld + k0 + hi * 8;
  F16Frag f;
  f.q[0] = *(const v4i*)(p);
  f.q[1] = *(const v4i*)(p + 16);
  return f.v;
}

// B 32x16 f16: lane<16 -> N=lane, VGPR0..3 K=0..7, VGPR4..7 K=16..23; lane>=16 -> K+8 / K+24.
// Source stored "N-major" (row n holds contiguous K) -> two contiguous b128.
__device__ __forceinline__ v16h load_b_frag(const _Float16* base,
                                            int n0, int ld, int k0) {
  const int lane = threadIdx.x & 31;
  const int hi   = lane >> 4;
  const int n    = lane & 15;
  const _Float16* p = base + (size_t)(n0 + n) * ld + k0 + hi * 8;
  F16Frag f;
  f.q[0] = *(const v4i*)(p);
  f.q[1] = *(const v4i*)(p + 16);
  return f.v;
}

// ---------------------------------------------------------------------------
// Tensor Data Mover: 2D f16 tile (tile_d1 rows x tile_d0 elems) -> LDS.
// D# built per ISA 8.3/8.4: group0 = {flags, lds_addr, global_addr, type=2},
// group1 = {data_size=2B, tensor dims/strides, tile dims}. 2D -> remaining
// descriptor groups zero. This toolchain's builtin takes 6 operands
// (uint32x4, int32x8, int32x4, int32x4, int32x8, i32 cpol).
#if __has_builtin(__builtin_amdgcn_tensor_load_to_lds) && __has_builtin(__builtin_amdgcn_s_wait_tensorcnt)
#define HAVE_TDM 1
__device__ __forceinline__ void tdm_load_tile_2d(unsigned lds_off, const void* gsrc,
                                                 unsigned tensor_d0, unsigned tensor_d1,
                                                 unsigned stride0,
                                                 unsigned tile_d0, unsigned tile_d1) {
  unsigned long long ga = (unsigned long long)gsrc;
  u32x4 g0;
  g0[0] = 1u;                                           // count=1, user D#
  g0[1] = lds_off;                                      // lds_addr (bytes)
  g0[2] = (unsigned)(ga & 0xFFFFFFFFu);                 // global_addr[31:0]
  g0[3] = (unsigned)((ga >> 32) & 0x01FFFFFFu) | (2u << 30);  // ga[56:32] | type=2
  i32x8 g1;
  g1[0] = (int)(1u << 16);                              // data_size=1 (2 bytes)
  g1[1] = (int)((tensor_d0 & 0xFFFFu) << 16);           // tensor_dim0[15:0]
  g1[2] = (int)(((tensor_d0 >> 16) & 0xFFFFu) |
                ((tensor_d1 & 0xFFFFu) << 16));         // dim0[31:16] | dim1[15:0]
  g1[3] = (int)(((tensor_d1 >> 16) & 0xFFFFu) |
                ((tile_d0 & 0xFFFFu) << 16));           // dim1[31:16] | tile_dim0
  g1[4] = (int)(tile_d1 & 0xFFFFu);                     // tile_dim1, tile_dim2=0
  g1[5] = (int)stride0;                                 // tensor_dim0_stride[31:0]
  g1[6] = 0;                                            // stride0[47:32], stride1 lo
  g1[7] = 0;
  v4i  gz4 = {0, 0, 0, 0};
  i32x8 gz8 = {0, 0, 0, 0, 0, 0, 0, 0};
  __builtin_amdgcn_tensor_load_to_lds(g0, g1, gz4, gz4, gz8, 0);
}
#else
#define HAVE_TDM 0
#endif

__device__ __forceinline__ unsigned lds_off_of(const void* p) {
  return (unsigned)(unsigned long long)p;   // generic LDS addr: offset in [31:0]
}

// ---------------------------------------------------------------------------
// GroupNorm: one block per (b, group). Writes normalized f16, TRANSPOSED to
// hT[B, HW, C] so GEMM B-fragments are contiguous b128 loads.
__global__ void gn_kernel(const float* __restrict__ x,
                          const float* __restrict__ gamma,
                          const float* __restrict__ beta,
                          _Float16* __restrict__ hT) {
  const int b = blockIdx.x >> 5;
  const int g = blockIdx.x & 31;
  const int cpg = CH / NGROUP;                 // 8 channels per group
  const float* xg = x + ((size_t)b * CH + g * cpg) * HWTOK;
  const int total = cpg * HWTOK;               // 32768

  float s = 0.f, ss = 0.f;
  for (int i = threadIdx.x; i < total; i += blockDim.x) {
    float v = xg[i];
    s += v; ss += v * v;
  }
  const int lane = threadIdx.x & 31;
  const int warp = threadIdx.x >> 5;
#pragma unroll
  for (int off = 16; off >= 1; off >>= 1) {
    s  += __shfl_xor(s,  off);
    ss += __shfl_xor(ss, off);
  }
  __shared__ float red[16];
  if (lane == 0) { red[warp] = s; red[8 + warp] = ss; }
  __syncthreads();
  float ts = 0.f, tss = 0.f;
#pragma unroll
  for (int w = 0; w < 8; ++w) { ts += red[w]; tss += red[8 + w]; }

  const float mean = ts * (1.0f / (float)total);
  const float var  = tss * (1.0f / (float)total) - mean * mean;
  const float rinv = rsqrtf(var + GNEPS);

  for (int i = threadIdx.x; i < total; i += blockDim.x) {
    int cl = i >> 12;
    int hw = i & (HWTOK - 1);
    int c  = g * cpg + cl;
    float v = (xg[i] - mean) * rinv * gamma[c] + beta[c];
    hT[((size_t)b * HWTOK + hw) * CH + c] = (_Float16)v;
  }
}

// ---------------------------------------------------------------------------
__global__ void cvt_w_kernel(const float* __restrict__ wqkv,
                             const float* __restrict__ wproj,
                             _Float16* __restrict__ wq16,
                             _Float16* __restrict__ wp16) {
  int i = blockIdx.x * blockDim.x + threadIdx.x;
  if (i < 3 * CH * CH) wq16[i] = (_Float16)wqkv[i];
  if (i < CH * CH)     wp16[i] = (_Float16)wproj[i];
}

// ---------------------------------------------------------------------------
// QKV GEMM, register-blocked 64(M) x 16(N) per wave: one activation B-fragment
// shared across 4 weight A-tiles -> 4 WMMAs per K-step, 32 per wave.
__global__ void qkv_gemm_kernel(const _Float16* __restrict__ hT,
                                const _Float16* __restrict__ w16,
                                const float* __restrict__ bias,
                                _Float16* __restrict__ qf,
                                _Float16* __restrict__ kf,
                                _Float16* __restrict__ vf) {
  const int wave = (blockIdx.x * blockDim.x + threadIdx.x) >> 5;
  const int b  = wave / (12 * 256);
  const int rm = wave % (12 * 256);
  const int mg = rm >> 8;            // 0..11  (64-wide o block)
  const int nt = rm & 255;           // 0..255 (hw tile)

  const _Float16* hTb = hT + (size_t)b * HWTOK * CH;
  v8f acc[4] = {};
#pragma unroll
  for (int kk = 0; kk < 8; ++kk) {
    v16h bf = load_b_frag(hTb, nt * 16, CH, kk * 32);
#pragma unroll
    for (int t = 0; t < 4; ++t) {
      v16h a = load_a_frag(w16, mg * 64 + t * 16, CH, kk * 32);
      acc[t] = __builtin_amdgcn_wmma_f32_16x16x32_f16(false, a, false, bf,
                                                      (short)0, acc[t], false, false);
    }
  }

  const int lane = threadIdx.x & 31;
  const int hi   = lane >> 4;
  const int hw   = nt * 16 + (lane & 15);
#pragma unroll
  for (int t = 0; t < 4; ++t) {
#pragma unroll
    for (int r = 0; r < 8; ++r) {
      int m = mg * 64 + t * 16 + r + 8 * hi;   // output channel o in [0,768)
      float val = acc[t][r] + bias[m];
      int seg  = m >> 8;                       // 0=q 1=k 2=v (wave-uniform)
      int oc   = m & 255;
      int head = oc >> 5;
      int d    = oc & 31;
      _Float16 h16 = (_Float16)val;
      if (seg == 0) {
        qf[(((size_t)b * NHEAD + head) * HWTOK + hw) * HDIM + d] = h16;
      } else if (seg == 1) {
        kf[(((size_t)b * NHEAD + head) * HWTOK + hw) * HDIM + d] = h16;
      } else {
        vf[(((size_t)b * NHEAD + head) * HDIM + d) * HWTOK + hw] = h16;
      }
    }
  }
}

// ---------------------------------------------------------------------------
// Flash attention. Block = 8 waves = 8 query tiles of the SAME (b, head):
// the 32-key K/V chunk is staged into LDS once per block (TDM tile DMA when
// available, cooperative b128 loads otherwise) and reused by all 8 waves.
__global__ void attn_kernel(const _Float16* __restrict__ qf,
                            const _Float16* __restrict__ kf,
                            const _Float16* __restrict__ vf,
                            _Float16* __restrict__ aout) {
  __shared__ __align__(16) _Float16 kstage[32 * 32];       // [key_local][d]
  __shared__ __align__(16) _Float16 vstage[32 * 32];       // [d][key_local]
  __shared__ __align__(16) _Float16 pstage[8 * 16 * 32];   // per-wave P tile

  const int warp = threadIdx.x >> 5;
  const int lane = threadIdx.x & 31;
  const int hi   = lane >> 4;
  const int n15  = lane & 15;

  const int wave = blockIdx.x * (blockDim.x >> 5) + warp;
  const int qt = wave & 255;
  const int h  = (wave >> 8) & 7;
  const int b  = wave >> 11;

  const _Float16* qb = qf + ((size_t)b * NHEAD + h) * HWTOK * HDIM;
  const _Float16* kb = kf + ((size_t)b * NHEAD + h) * HWTOK * HDIM;
  const _Float16* vb = vf + ((size_t)b * NHEAD + h) * HDIM * HWTOK;
  _Float16* pw = &pstage[warp * 16 * 32];

  const v16h qa = load_a_frag(qb, qt * 16, HDIM, 0);   // full d=32 per tile

  v8f o0 = {}, o1 = {};
  float mrow[8], lrow[8];
#pragma unroll
  for (int r = 0; r < 8; ++r) { mrow[r] = -1e30f; lrow[r] = 0.f; }
  const float scale = 0.1767766952966369f;             // 1/sqrt(32)

  for (int kc = 0; kc < HWTOK; kc += 32) {
    __syncthreads();   // previous chunk fully consumed before restage
#if HAVE_TDM
    if (warp == 0) {
      // K tile: tensor [4096 x 32] f16, tile rows kc..kc+31, stride 32
      tdm_load_tile_2d(lds_off_of(kstage), kb + (size_t)kc * HDIM,
                       HDIM, HWTOK, HDIM, HDIM, 32);
      // V tile: tensor [32 x 4096] f16, tile cols kc..kc+31, stride 4096
      tdm_load_tile_2d(lds_off_of(vstage), vb + kc,
                       HWTOK, HDIM, HWTOK, 32, HDIM);
      __builtin_amdgcn_s_wait_tensorcnt(0);
    }
#else
    {
      int t = threadIdx.x;
      if (t < 128) {          // K tile: 32 rows x 4 x b128
        int key = t >> 2, ch = t & 3;
        ((v4i*)kstage)[t] = *(const v4i*)(kb + (size_t)(kc + key) * HDIM + ch * 8);
      } else {                // V tile: 32 d-rows x 4 x b128
        int i = t - 128;
        int d = i >> 2, ch = i & 3;
        ((v4i*)vstage)[i] = *(const v4i*)(vb + (size_t)d * HWTOK + kc + ch * 8);
      }
    }
#endif
    __syncthreads();   // tiles visible to all waves

    // S = Q x K^T from LDS K tile
    v16h kb0 = load_b_frag(kstage, 0,  HDIM, 0);
    v16h kb1 = load_b_frag(kstage, 16, HDIM, 0);
    v8f z = {};
    v8f s0 = __builtin_amdgcn_wmma_f32_16x16x32_f16(false, qa, false, kb0,
                                                    (short)0, z, false, false);
    v8f s1 = __builtin_amdgcn_wmma_f32_16x16x32_f16(false, qa, false, kb1,
                                                    (short)0, z, false, false);
    float p0[8], p1[8];
#pragma unroll
    for (int r = 0; r < 8; ++r) { p0[r] = s0[r] * scale; p1[r] = s1[r] * scale; }

    // online softmax (16 lanes share each row)
#pragma unroll
    for (int r = 0; r < 8; ++r) {
      float m2 = fmaxf(p0[r], p1[r]);
#pragma unroll
      for (int off = 1; off < 16; off <<= 1) m2 = fmaxf(m2, __shfl_xor(m2, off));
      float mn = fmaxf(mrow[r], m2);
      float alpha = __expf(mrow[r] - mn);
      p0[r] = __expf(p0[r] - mn);
      p1[r] = __expf(p1[r] - mn);
      float su = p0[r] + p1[r];
#pragma unroll
      for (int off = 1; off < 16; off <<= 1) su += __shfl_xor(su, off);
      lrow[r] = lrow[r] * alpha + su;
      o0[r] *= alpha;
      o1[r] *= alpha;
      mrow[r] = mn;
    }
    // P: C layout -> LDS row-major [q][key] -> reload as A fragment
#pragma unroll
    for (int r = 0; r < 8; ++r) {
      int row = r + 8 * hi;
      pw[row * 32 + n15]      = (_Float16)p0[r];
      pw[row * 32 + 16 + n15] = (_Float16)p1[r];
    }
    F16Frag pf;
    const _Float16* pp = pw + n15 * 32 + hi * 8;
    pf.q[0] = *(const v4i*)(pp);
    pf.q[1] = *(const v4i*)(pp + 16);

    // O += P x V from LDS V tile (two d-halves)
    v16h vb0 = load_b_frag(vstage, 0,  32, 0);
    v16h vb1 = load_b_frag(vstage, 16, 32, 0);
    o0 = __builtin_amdgcn_wmma_f32_16x16x32_f16(false, pf.v, false, vb0,
                                                (short)0, o0, false, false);
    o1 = __builtin_amdgcn_wmma_f32_16x16x32_f16(false, pf.v, false, vb1,
                                                (short)0, o1, false, false);
  }

  // normalize and store to aout[B, HW, C] (proj-GEMM friendly)
#pragma unroll
  for (int r = 0; r < 8; ++r) {
    int row = qt * 16 + r + 8 * hi;
    float invl = 1.0f / lrow[r];
    size_t base = ((size_t)b * HWTOK + row) * CH + h * HDIM;
    aout[base + n15]      = (_Float16)(o0[r] * invl);
    aout[base + 16 + n15] = (_Float16)(o1[r] * invl);
  }
}

// ---------------------------------------------------------------------------
// Proj GEMM (64x16 per wave) + bias + f32 residual.
__global__ void proj_kernel(const _Float16* __restrict__ aout,
                            const _Float16* __restrict__ wp16,
                            const float* __restrict__ bproj,
                            const float* __restrict__ x,
                            float* __restrict__ out) {
  const int wave = (blockIdx.x * blockDim.x + threadIdx.x) >> 5;
  const int b  = wave >> 10;
  const int rm = wave & 1023;
  const int mg = rm >> 8;            // 0..3 (64-wide o block)
  const int nt = rm & 255;

  const _Float16* ab = aout + (size_t)b * HWTOK * CH;
  v8f acc[4] = {};
#pragma unroll
  for (int kk = 0; kk < 8; ++kk) {
    v16h bf = load_b_frag(ab, nt * 16, CH, kk * 32);
#pragma unroll
    for (int t = 0; t < 4; ++t) {
      v16h a = load_a_frag(wp16, mg * 64 + t * 16, CH, kk * 32);
      acc[t] = __builtin_amdgcn_wmma_f32_16x16x32_f16(false, a, false, bf,
                                                      (short)0, acc[t], false, false);
    }
  }
  const int lane = threadIdx.x & 31;
  const int hi   = lane >> 4;
  const int hw   = nt * 16 + (lane & 15);
#pragma unroll
  for (int t = 0; t < 4; ++t) {
#pragma unroll
    for (int r = 0; r < 8; ++r) {
      int m = mg * 64 + t * 16 + r + 8 * hi;
      size_t idx = ((size_t)b * CH + m) * HWTOK + hw;
      out[idx] = x[idx] + acc[t][r] + bproj[m];
    }
  }
}

// ---------------------------------------------------------------------------
extern "C" void kernel_launch(void* const* d_in, const int* in_sizes, int n_in,
                              void* d_out, int out_size, void* d_ws, size_t ws_size,
                              hipStream_t stream) {
  const float* x     = (const float*)d_in[0];
  const float* wqkv  = (const float*)d_in[1];
  const float* bqkv  = (const float*)d_in[2];
  const float* wproj = (const float*)d_in[3];
  const float* bproj = (const float*)d_in[4];
  const float* gam   = (const float*)d_in[5];
  const float* bet   = (const float*)d_in[6];
  float* out = (float*)d_out;

  char* ws = (char*)d_ws;
  size_t off = 0;
  auto carve = [&](size_t bytes) {
    void* p = ws + off;
    off += (bytes + 255) & ~(size_t)255;
    return p;
  };
  _Float16* hT    = (_Float16*)carve((size_t)BATCH * HWTOK * CH * 2);
  _Float16* qf    = (_Float16*)carve((size_t)BATCH * NHEAD * HWTOK * HDIM * 2);
  _Float16* kf    = (_Float16*)carve((size_t)BATCH * NHEAD * HWTOK * HDIM * 2);
  _Float16* vf    = (_Float16*)carve((size_t)BATCH * NHEAD * HDIM * HWTOK * 2);
  _Float16* aoutb = (_Float16*)carve((size_t)BATCH * HWTOK * CH * 2);
  _Float16* wq16  = (_Float16*)carve((size_t)3 * CH * CH * 2);
  _Float16* wp16  = (_Float16*)carve((size_t)CH * CH * 2);

  gn_kernel<<<BATCH * NGROUP, 256, 0, stream>>>(x, gam, bet, hT);
  cvt_w_kernel<<<(3 * CH * CH + 255) / 256, 256, 0, stream>>>(wqkv, wproj, wq16, wp16);
  // QKV: 6144 tile-waves (64x16 each), 8 waves/block
  qkv_gemm_kernel<<<(BATCH * 12 * 256) / 8, 256, 0, stream>>>(hT, wq16, bqkv, qf, kf, vf);
  // attention: 4096 q-tile waves, 8 per block sharing K/V through LDS
  attn_kernel<<<(BATCH * NHEAD * (HWTOK / 16)) / 8, 256, 0, stream>>>(qf, kf, vf, aoutb);
  // proj: 2048 tile-waves (64x16 each)
  proj_kernel<<<(BATCH * 4 * 256) / 8, 256, 0, stream>>>(aoutb, wp16, bproj, x, out);
}
